// RecurrentLayer_730144441066
// MI455X (gfx1250) — matrix operand order repaired and checked by reading, not compiled, not run on hardware
//
#include <hip/hip_runtime.h>
#include <hip/hip_bf16.h>

typedef __attribute__((ext_vector_type(16))) _Float16 v16h;
typedef __attribute__((ext_vector_type(8)))  float    v8f;

#define TT 1024
#define BB 64
#define DD 512
#define HH 512
#define H4 2048    // 4*H
#define KTOT 1024  // D + H
#define NT_PER_GATE 32   // H/16
#define KSTEPS_X 16      // D/32
#define KSTEPS_H 16      // H/32

// ---------------------------------------------------------------------------
// WMMA f16 fragment maps (CDNA5 ISA 7.12.2, wave32):
//  A (16x32):  lane l -> M=l&15; khalf=8*(l>>4); slot s:
//              s in [0,8)  -> K = khalf + s
//              s in [8,16) -> K = 16 + khalf + (s-8)
//  B (32x16):  lane l -> N=l&15; K map same as A (khalf=8*(l>>4), same slots)
//  C/D (16x16 f32): lane l -> N=l&15; reg r -> M = r + 8*(l>>4)
// Packed buffers store each wave-fragment as lane-major 32B runs so the hot
// loop issues pure contiguous b128 loads.
// ---------------------------------------------------------------------------

// Pack [Wx; Wh] fp32 [K=1024, N=2048] -> f16 B-fragment order:
// idx = ((kt*128 + nt)*32 + lane)*16 + s
__global__ void lstm_pack_w(const float* __restrict__ Wx,
                            const float* __restrict__ Wh,
                            _Float16* __restrict__ pw) {
    int idx = blockIdx.x * blockDim.x + threadIdx.x;
    if (idx >= KTOT * H4) return;
    int s    = idx & 15;
    int l    = (idx >> 4) & 31;
    int tile = idx >> 9;
    int nt   = tile & 127;
    int kt   = tile >> 7;
    int k    = kt * 32 + ((s < 8) ? ((l >> 4) * 8 + s) : (16 + (l >> 4) * 8 + (s - 8)));
    int n    = nt * 16 + (l & 15);
    float w  = (k < DD) ? Wx[(size_t)k * H4 + n]
                        : Wh[(size_t)(k - DD) * H4 + n];
    pw[idx] = (_Float16)w;
}

// Pack x fp32 [T,B,D] -> f16 A-fragment order:
// idx = (((t*4 + m_tile)*16 + ks)*32 + lane)*16 + s
__global__ void lstm_pack_x(const float* __restrict__ x,
                            _Float16* __restrict__ px) {
    long long idx = (long long)blockIdx.x * blockDim.x + threadIdx.x;
    if (idx >= (long long)TT * BB * DD) return;
    int s      = (int)(idx & 15);
    int l      = (int)((idx >> 4) & 31);
    long long rest = idx >> 9;
    int ks     = (int)(rest & 15);
    int m_tile = (int)((rest >> 4) & 3);
    long long t = rest >> 6;
    int khalf  = 8 * (l >> 4);
    int k_local = (s < 8) ? (khalf + s) : (16 + khalf + (s - 8));
    int k      = ks * 32 + k_local;
    int row    = m_tile * 16 + (l & 15);
    px[idx] = (_Float16)x[(t * BB + row) * DD + k];
}

// Pack h0 fp32 [B,H] -> f16 A-fragment order (same map, 16 ks tiles).
__device__ __forceinline__ int h_pack_index(int row, int j) {
    int ks  = j >> 5;
    int kl  = j & 31;
    int sub = kl & 15;
    int hi_a = sub >> 3;
    int off  = sub & 7;
    int s    = (kl < 16) ? off : (8 + off);
    int m_tile = row >> 4;
    int m      = row & 15;
    return (((m_tile * 16 + ks) * 32) + hi_a * 16 + m) * 16 + s;
}

__global__ void lstm_pack_h0(const float* __restrict__ h0,
                             _Float16* __restrict__ hb) {
    int idx = blockIdx.x * blockDim.x + threadIdx.x;
    if (idx >= BB * HH) return;
    int row = idx / HH, j = idx % HH;
    hb[h_pack_index(row, j)] = (_Float16)h0[idx];
}

__global__ void lstm_init_c(const float* __restrict__ c0,
                            float* __restrict__ cbuf) {
    int idx = blockIdx.x * blockDim.x + threadIdx.x;
    if (idx < BB * HH) cbuf[idx] = c0[idx];
}

__device__ __forceinline__ float sigmoidf_(float v) {
    return 1.0f / (1.0f + __expf(-v));
}
// Saturation-safe fast tanh: 1 - 2/(e^{2x}+1); exp->inf gives exactly 1.
__device__ __forceinline__ float tanhf_(float v) {
    return 1.0f - 2.0f / (__expf(2.0f * v) + 1.0f);
}

// ---------------------------------------------------------------------------
// One timestep. 16 blocks x 256 threads = 128 waves.
// m_tile = gid&3 (batch rows), j_tile = gid>>2 (hidden-column slice).
// Each wave: 4 accumulators (gates i,f,g,o over same columns), K-loop of
// pure f16 fragment loads + v_wmma, then fully in-register LSTM update.
// Writes h both as fp32 (h_seq output) and f16 A-fragments (next step).
// ---------------------------------------------------------------------------
__global__ void __launch_bounds__(256)
lstm_step(const _Float16* __restrict__ px,     // packed x fragments
          const _Float16* __restrict__ hread,  // packed h_{t-1} fragments
          _Float16* __restrict__ hwrite,       // packed h_t fragments
          const _Float16* __restrict__ pw,     // packed W fragments
          const float* __restrict__ bias,      // [4H]
          float* __restrict__ cbuf,            // [B,H] cell state
          float* __restrict__ out,             // h_seq + hT + cT
          int t, int tn) {                     // tn = min(t+1, T-1)
    const int lane = threadIdx.x & 31;
    const int wave = threadIdx.x >> 5;
    const int gid  = blockIdx.x * 8 + wave;
    const int m_tile = gid & 3;
    const int j_tile = gid >> 2;

    const v16h* pax  = (const v16h*)px +
                       ((size_t)t  * 4 + m_tile) * (KSTEPS_X * 32);
    const v16h* paxn = (const v16h*)px +
                       ((size_t)tn * 4 + m_tile) * (KSTEPS_X * 32);
    const v16h* pah  = (const v16h*)hread + (size_t)m_tile * (KSTEPS_H * 32);
    const v16h* pw32 = (const v16h*)pw;

    v8f acc0 = {}, acc1 = {}, acc2 = {}, acc3 = {};

    // ---- x half of K (kt = 0..15); prefetch next step's x fragments.
#pragma unroll 4
    for (int ks = 0; ks < KSTEPS_X; ++ks) {
        __builtin_prefetch((const void*)(paxn + ks * 32 + lane), 0, 1);
        v16h a = pax[ks * 32 + lane];
        const size_t base = (size_t)ks * 128 * 32 + lane;
        v16h b0 = pw32[base + (size_t)(j_tile + 0 * NT_PER_GATE) * 32];
        v16h b1 = pw32[base + (size_t)(j_tile + 1 * NT_PER_GATE) * 32];
        v16h b2 = pw32[base + (size_t)(j_tile + 2 * NT_PER_GATE) * 32];
        v16h b3 = pw32[base + (size_t)(j_tile + 3 * NT_PER_GATE) * 32];
        acc0 = __builtin_amdgcn_wmma_f32_16x16x32_f16(false, a, false, b0,
                                                      (short)0, acc0, false, false);
        acc1 = __builtin_amdgcn_wmma_f32_16x16x32_f16(false, a, false, b1,
                                                      (short)0, acc1, false, false);
        acc2 = __builtin_amdgcn_wmma_f32_16x16x32_f16(false, a, false, b2,
                                                      (short)0, acc2, false, false);
        acc3 = __builtin_amdgcn_wmma_f32_16x16x32_f16(false, a, false, b3,
                                                      (short)0, acc3, false, false);
    }
    // ---- h half of K (kt = 16..31).
#pragma unroll 4
    for (int ks = 0; ks < KSTEPS_H; ++ks) {
        v16h a = pah[ks * 32 + lane];
        const size_t base = (size_t)(ks + KSTEPS_X) * 128 * 32 + lane;
        v16h b0 = pw32[base + (size_t)(j_tile + 0 * NT_PER_GATE) * 32];
        v16h b1 = pw32[base + (size_t)(j_tile + 1 * NT_PER_GATE) * 32];
        v16h b2 = pw32[base + (size_t)(j_tile + 2 * NT_PER_GATE) * 32];
        v16h b3 = pw32[base + (size_t)(j_tile + 3 * NT_PER_GATE) * 32];
        acc0 = __builtin_amdgcn_wmma_f32_16x16x32_f16(false, a, false, b0,
                                                      (short)0, acc0, false, false);
        acc1 = __builtin_amdgcn_wmma_f32_16x16x32_f16(false, a, false, b1,
                                                      (short)0, acc1, false, false);
        acc2 = __builtin_amdgcn_wmma_f32_16x16x32_f16(false, a, false, b2,
                                                      (short)0, acc2, false, false);
        acc3 = __builtin_amdgcn_wmma_f32_16x16x32_f16(false, a, false, b3,
                                                      (short)0, acc3, false, false);
    }

    // ---- Fused LSTM epilogue (C/D layout: N=lane&15, M=r+8*(lane>>4)).
    const int hi_c = lane >> 4;
    const int j    = j_tile * 16 + (lane & 15);
    const float bi = bias[0 * HH + j];
    const float bf = bias[1 * HH + j];
    const float bg = bias[2 * HH + j];
    const float bo = bias[3 * HH + j];

    // Per-lane constants of the packed-h destination map (j fixed per lane).
    const int ks_h = j >> 5;
    const int kl   = j & 31;
    const int sub  = kl & 15;
    const int hi_a = sub >> 3;
    const int off  = sub & 7;
    const int s_h  = (kl < 16) ? off : (8 + off);
    const int hbase = (((m_tile * 16 + ks_h) * 32) + hi_a * 16) * 16 + s_h;

    float* outh = out + (size_t)t * BB * HH;

#pragma unroll
    for (int r = 0; r < 8; ++r) {
        const int m    = r + 8 * hi_c;
        const int brow = m_tile * 16 + m;
        const size_t o = (size_t)brow * HH + j;
        const float gi = acc0[r] + bi;
        const float gf = acc1[r] + bf;
        const float gg = acc2[r] + bg;
        const float go = acc3[r] + bo;
        const float c_new = sigmoidf_(gf) * cbuf[o] + sigmoidf_(gi) * tanhf_(gg);
        const float h_new = sigmoidf_(go) * tanhf_(c_new);
        cbuf[o] = c_new;
        outh[o] = h_new;
        hwrite[hbase + m * 16] = (_Float16)h_new;   // next step's A fragment
    }
}

// hT (= h_seq[T-1]) and cT appended to d_out in tuple return order.
__global__ void lstm_finalize(float* __restrict__ out,
                              const float* __restrict__ cbuf) {
    int idx = blockIdx.x * blockDim.x + threadIdx.x;
    if (idx < BB * HH) {
        out[(size_t)TT * BB * HH + idx]           = out[(size_t)(TT - 1) * BB * HH + idx];
        out[(size_t)TT * BB * HH + BB * HH + idx] = cbuf[idx];
    }
}

extern "C" void kernel_launch(void* const* d_in, const int* in_sizes, int n_in,
                              void* d_out, int out_size, void* d_ws, size_t ws_size,
                              hipStream_t stream) {
    const float* x  = (const float*)d_in[0];
    const float* h0 = (const float*)d_in[1];
    const float* c0 = (const float*)d_in[2];
    const float* Wx = (const float*)d_in[3];
    const float* Wh = (const float*)d_in[4];
    const float* b  = (const float*)d_in[5];
    float* out = (float*)d_out;

    char* ws = (char*)d_ws;
    _Float16* pw  = (_Float16*)ws;                       // 4 MB
    ws += (size_t)KTOT * H4 * sizeof(_Float16);
    _Float16* px  = (_Float16*)ws;                       // 64 MB
    ws += (size_t)TT * BB * DD * sizeof(_Float16);
    _Float16* hb0 = (_Float16*)ws;                       // 64 KB
    ws += (size_t)BB * HH * sizeof(_Float16);
    _Float16* hb1 = (_Float16*)ws;                       // 64 KB
    ws += (size_t)BB * HH * sizeof(_Float16);
    float* cbf = (float*)ws;                             // 128 KB

    {   // One-time (per launch) packing + state init.
        int nw = KTOT * H4;
        lstm_pack_w<<<(nw + 255) / 256, 256, 0, stream>>>(Wx, Wh, pw);
        long long nx = (long long)TT * BB * DD;
        lstm_pack_x<<<(unsigned)((nx + 255) / 256), 256, 0, stream>>>(x, px);
        lstm_pack_h0<<<(BB * HH + 255) / 256, 256, 0, stream>>>(h0, hb0);
        lstm_init_c<<<(BB * HH + 255) / 256, 256, 0, stream>>>(c0, cbf);
    }

    // Serial recurrence; h fragments ping-pong between hb0/hb1.
    for (int t = 0; t < TT; ++t) {
        const _Float16* hr = (t & 1) ? hb1 : hb0;
        _Float16*       hw = (t & 1) ? hb0 : hb1;
        int tn = (t + 1 < TT) ? (t + 1) : t;
        lstm_step<<<16, 256, 0, stream>>>(px, hr, hw, pw, b, cbf, out, t, tn);
    }

    lstm_finalize<<<(BB * HH + 255) / 256, 256, 0, stream>>>(out, cbf);
}
